// CrossAttention_44212393345593
// MI455X (gfx1250) — matrix operand order repaired and checked by reading, not compiled, not run on hardware
//
#include <hip/hip_runtime.h>
#include <hip/hip_bf16.h>

// ---------------------------------------------------------------------------
// CDNA5 (gfx1250) cross-attention, wave32 + v_wmma_f32_16x16x32_f16 everywhere.
// ---------------------------------------------------------------------------

typedef __attribute__((ext_vector_type(16))) _Float16 v16h;
typedef __attribute__((ext_vector_type(8)))  _Float16 v8h;
typedef __attribute__((ext_vector_type(8)))  float    v8f;

#define B_  4
#define S_  2048
#define SC_ 2048
#define DIM_ 1024
#define H_  16
#define HD_ 64

// ------------------------- fp32 -> fp16 convert ----------------------------
__global__ __launch_bounds__(256) void ca_f32_to_f16(const float* __restrict__ a,
                                                     _Float16* __restrict__ b) {
  size_t i = (size_t)blockIdx.x * 256 + threadIdx.x;
  b[i] = (_Float16)a[i];
}

// ------------------------- generic WMMA GEMM -------------------------------
// C[M,N] (f32) = A[M,K] (f16 row-major) * Bm[K,N] (f16 row-major) (+ bias)
// Workgroup tile 128x128, 8 waves (2x4); each wave 64x32 = 4x2 WMMA tiles
// -> 8 v_wmma per K-step against 6 fragment loads.
#define GBM 128
#define GBN 128
#define GTK 32
#define ASTR 40   // halves; row pitch 80B keeps 16B alignment for v8h ds loads

__global__ __launch_bounds__(256) void ca_gemm_f16(
    const _Float16* __restrict__ A, const _Float16* __restrict__ Bm,
    float* __restrict__ C, const float* __restrict__ bias,
    int M, int N, int K) {
  __shared__ _Float16 As[GBM * ASTR];   // 10 KB
  __shared__ _Float16 Bs[GBN * ASTR];   // 10 KB, stored transposed: Bs[n][k]
  const int tid  = threadIdx.x;
  const int lane = tid & 31;
  const int wave = tid >> 5;
  const int waveM = wave >> 2;          // 0..1  (64 rows each)
  const int waveN = wave & 3;           // 0..3  (32 cols each)
  const int bm = blockIdx.x * GBM;
  const int bn = blockIdx.y * GBN;
  const int l16 = lane & 15;
  const int h8  = (lane < 16) ? 0 : 8;  // A-frag K offset / C row offset
  const int h16 = (lane < 16) ? 0 : 16; // unused; B-frag K offset is h8 as well

  // staging assignments (loop-invariant)
  const int arow0 = tid >> 2,            akc0 = (tid & 3) << 3;
  const int arow1 = (tid + 256) >> 2,    akc1 = ((tid + 256) & 3) << 3;
  const int bk0   = tid >> 4,            bn0  = (tid & 15) << 3;
  const int bk1   = (tid + 256) >> 4,    bn1  = ((tid + 256) & 15) << 3;

  // strength-reduced tile pointers (no per-iteration 64-bit multiplies)
  const _Float16* ap0 = A + (size_t)(bm + arow0) * K + akc0;
  const _Float16* ap1 = A + (size_t)(bm + arow1) * K + akc1;
  const _Float16* bp0 = Bm + (size_t)bk0 * N + bn + bn0;
  const _Float16* bp1 = Bm + (size_t)bk1 * N + bn + bn1;
  const size_t bstep = (size_t)GTK * N;

  v8f acc[4][2] = {};

  for (int k0 = 0; k0 < K; k0 += GTK) {
    // issue ALL staging loads first (one clause, one loadcnt wait), then store
    v8h a0 = *(const v8h*)ap0;
    v8h a1 = *(const v8h*)ap1;
    v8h b0 = *(const v8h*)bp0;
    v8h b1 = *(const v8h*)bp1;
    // prefetch next K-tile into cache while this one is consumed
    if (k0 + GTK < K) {
      __builtin_prefetch(ap0 + GTK, 0, 1);
      __builtin_prefetch(bp0 + bstep, 0, 1);
    }
    ap0 += GTK; ap1 += GTK; bp0 += bstep; bp1 += bstep;

    *(v8h*)&As[arow0 * ASTR + akc0] = a0;
    *(v8h*)&As[arow1 * ASTR + akc1] = a1;
#pragma unroll
    for (int j = 0; j < 8; ++j) Bs[(bn0 + j) * ASTR + bk0] = b0[j];
#pragma unroll
    for (int j = 0; j < 8; ++j) Bs[(bn1 + j) * ASTR + bk1] = b1[j];
    __syncthreads();

    v16h af[4], bf[2];
#pragma unroll
    for (int tm = 0; tm < 4; ++tm) {
      const _Float16* p = &As[(waveM * 64 + tm * 16 + l16) * ASTR + h8];
      v8h lo = *(const v8h*)p;
      v8h hi = *(const v8h*)(p + 16);
#pragma unroll
      for (int i = 0; i < 8; ++i) { af[tm][i] = lo[i]; af[tm][8 + i] = hi[i]; }
    }
#pragma unroll
    for (int tn = 0; tn < 2; ++tn) {
      const _Float16* p = &Bs[(waveN * 32 + tn * 16 + l16) * ASTR + h8];
      v8h lo = *(const v8h*)p;
      v8h hi = *(const v8h*)(p + 16);
#pragma unroll
      for (int i = 0; i < 8; ++i) { bf[tn][i] = lo[i]; bf[tn][8 + i] = hi[i]; }
    }
#pragma unroll
    for (int tm = 0; tm < 4; ++tm)
#pragma unroll
      for (int tn = 0; tn < 2; ++tn)
        acc[tm][tn] = __builtin_amdgcn_wmma_f32_16x16x32_f16(
            false, af[tm], false, bf[tn], (short)0, acc[tm][tn], false, false);
    __syncthreads();
  }
  // epilogue: C layout = lanes 0-15 rows i, lanes 16-31 rows i+8
#pragma unroll
  for (int tn = 0; tn < 2; ++tn) {
    const int col = bn + waveN * 32 + tn * 16 + l16;
    const float bb = bias ? bias[col] : 0.f;
#pragma unroll
    for (int tm = 0; tm < 4; ++tm) {
      float* cp = &C[(size_t)(bm + waveM * 64 + tm * 16 + h8) * N + col];
#pragma unroll
      for (int i = 0; i < 8; ++i) cp[(size_t)i * N] = acc[tm][tn][i] + bb;
    }
  }
}

// -------------------- per-head LayerNorm + RoPE (q) ------------------------
// one wave per (b,s,h) row of 64; lane t owns the rotate-half pair (t, t+32)
__global__ __launch_bounds__(256) void ca_ln_rope_q(
    const float* __restrict__ qf, const float* __restrict__ g,
    const float* __restrict__ bta, _Float16* __restrict__ qh) {
  int w = blockIdx.x * 8 + (threadIdx.x >> 5);
  int t = threadIdx.x & 31;
  int h = w % H_;
  int s = (w / H_) % S_;
  int b = w / (H_ * S_);
  const float* src = qf + ((size_t)(b * S_ + s) * H_ + h) * HD_;
  float x1 = src[t], x2 = src[t + 32];
  float sum = x1 + x2, sq = x1 * x1 + x2 * x2;
#pragma unroll
  for (int m = 16; m >= 1; m >>= 1) {
    sum += __shfl_xor(sum, m, 32);
    sq  += __shfl_xor(sq,  m, 32);
  }
  float mu = sum * (1.f / 64.f);
  float var = sq * (1.f / 64.f) - mu * mu;
  float rstd = rsqrtf(var + 1e-5f);
  float y1 = (x1 - mu) * rstd * g[t]      + bta[t];
  float y2 = (x2 - mu) * rstd * g[t + 32] + bta[t + 32];
  // inv_freq = 10000^(-t/32)
  float invf = __expf(-(float)t * (9.210340371976184f / 32.f));
  float ang = (float)s * invf;
  float sn, cs;
  __sincosf(ang, &sn, &cs);
  _Float16* dst = qh + ((size_t)(b * H_ + h) * S_ + s) * HD_;
  dst[t]      = (_Float16)(y1 * cs - y2 * sn);
  dst[t + 32] = (_Float16)(y1 * sn + y2 * cs);
}

// --------------- per-head LayerNorm (k) + transpose-store (v) --------------
__global__ __launch_bounds__(256) void ca_ln_kv(
    const float* __restrict__ kvf, const float* __restrict__ g,
    const float* __restrict__ bta, _Float16* __restrict__ kh,
    _Float16* __restrict__ vt) {
  int w = blockIdx.x * 8 + (threadIdx.x >> 5);
  int t = threadIdx.x & 31;
  int h = w % H_;
  int s = (w / H_) % SC_;
  int b = w / (H_ * SC_);
  const float* ks = kvf + (((size_t)(b * SC_ + s) * 2 + 0) * H_ + h) * HD_;
  const float* vs = kvf + (((size_t)(b * SC_ + s) * 2 + 1) * H_ + h) * HD_;
  float x1 = ks[t], x2 = ks[t + 32];
  float sum = x1 + x2, sq = x1 * x1 + x2 * x2;
#pragma unroll
  for (int m = 16; m >= 1; m >>= 1) {
    sum += __shfl_xor(sum, m, 32);
    sq  += __shfl_xor(sq,  m, 32);
  }
  float mu = sum * (1.f / 64.f);
  float rstd = rsqrtf(sq * (1.f / 64.f) - mu * mu + 1e-5f);
  _Float16* kd = kh + ((size_t)(b * H_ + h) * SC_ + s) * HD_;
  kd[t]      = (_Float16)((x1 - mu) * rstd * g[t]      + bta[t]);
  kd[t + 32] = (_Float16)((x2 - mu) * rstd * g[t + 32] + bta[t + 32]);
  // v stored transposed [B,H,Hd,Sc] so PV B-fragments are K-contiguous
  _Float16* vd = vt + ((size_t)(b * H_ + h) * HD_) * (size_t)SC_;
  vd[(size_t)t * SC_ + s]        = (_Float16)vs[t];
  vd[(size_t)(t + 32) * SC_ + s] = (_Float16)vs[t + 32];
}

// ----------------------- flash attention (per wave) ------------------------
// one wave owns a 16-row q tile; online softmax over 32 k-rows per iteration.
#define PSTR 40  // LDS P-tile pitch in halves (80B, keeps v8h alignment)

__global__ __launch_bounds__(256) void ca_flash_attn(
    const _Float16* __restrict__ qh, const _Float16* __restrict__ kh,
    const _Float16* __restrict__ vt, _Float16* __restrict__ oh) {
  __shared__ _Float16 Pl[8 * 16 * PSTR];
  const int tid = threadIdx.x, lane = tid & 31, wave = tid >> 5;
  _Float16* lp = &Pl[wave * 16 * PSTR];
  const int task = blockIdx.x * 8 + wave;   // B*H*(S/16) tasks
  const int qt = task % (S_ / 16);
  const int bh = task / (S_ / 16);
  const int b = bh / H_, h = bh % H_;
  const int l16 = lane & 15;
  const int h8  = (lane < 16) ? 0 : 8;
  const int h16 = (lane < 16) ? 0 : 16;

  // q A-fragments (K = 0..31 and 32..63 of Hd)
  const _Float16* qb = qh + ((size_t)bh * S_ + qt * 16) * HD_;
  v16h qf[2];
#pragma unroll
  for (int c = 0; c < 2; ++c) {
    const _Float16* p = qb + (size_t)l16 * HD_ + c * 32 + h8;
    v8h lo = *(const v8h*)p;
    v8h hi = *(const v8h*)(p + 16);
#pragma unroll
    for (int i = 0; i < 8; ++i) { qf[c][i] = lo[i]; qf[c][8 + i] = hi[i]; }
  }

  float mrow[8], lrow[8];
#pragma unroll
  for (int i = 0; i < 8; ++i) { mrow[i] = -1e30f; lrow[i] = 0.f; }
  v8f o[4] = {};

  const _Float16* kb = kh + (size_t)bh * SC_ * HD_;
  const _Float16* vb = vt + (size_t)bh * HD_ * (size_t)SC_;

  for (int kt = 0; kt < SC_ / 32; ++kt) {
    // S = q @ k^T : k rows act as B columns; k row-major is already K-contig
    v8f s0 = {}, s1 = {};
#pragma unroll
    for (int c = 0; c < 2; ++c) {
      v16h bk;
      {
        const _Float16* p = kb + (size_t)(kt * 32 + l16) * HD_ + c * 32 + h8;
        v8h lo = *(const v8h*)p;
        v8h hi = *(const v8h*)(p + 16);
#pragma unroll
        for (int i = 0; i < 8; ++i) { bk[i] = lo[i]; bk[8 + i] = hi[i]; }
      }
      s0 = __builtin_amdgcn_wmma_f32_16x16x32_f16(false, qf[c], false, bk,
                                                  (short)0, s0, false, false);
      {
        const _Float16* p = kb + (size_t)(kt * 32 + 16 + l16) * HD_ + c * 32 + h8;
        v8h lo = *(const v8h*)p;
        v8h hi = *(const v8h*)(p + 16);
#pragma unroll
        for (int i = 0; i < 8; ++i) { bk[i] = lo[i]; bk[8 + i] = hi[i]; }
      }
      s1 = __builtin_amdgcn_wmma_f32_16x16x32_f16(false, qf[c], false, bk,
                                                  (short)0, s1, false, false);
    }
    // online softmax: row r of the C tile lives in VGPR (r&7) of one 16-lane
    // half, so row reductions are shfl_xor with width 16.
    const float sc = 0.125f;  // 1/sqrt(64)
#pragma unroll
    for (int i = 0; i < 8; ++i) {
      float a0 = s0[i] * sc, a1 = s1[i] * sc;
      float mx = fmaxf(a0, a1);
#pragma unroll
      for (int m = 8; m >= 1; m >>= 1) mx = fmaxf(mx, __shfl_xor(mx, m, 16));
      float mn = fmaxf(mrow[i], mx);
      float alpha = __expf(mrow[i] - mn);
      float p0 = __expf(a0 - mn), p1 = __expf(a1 - mn);
      float rs = p0 + p1;
#pragma unroll
      for (int m = 8; m >= 1; m >>= 1) rs += __shfl_xor(rs, m, 16);
      lrow[i] = lrow[i] * alpha + rs;
      mrow[i] = mn;
#pragma unroll
      for (int j = 0; j < 4; ++j) o[j][i] *= alpha;
      int row = i + h8;
      lp[row * PSTR + l16]      = (_Float16)p0;
      lp[row * PSTR + 16 + l16] = (_Float16)p1;
    }
    // re-read P in A-fragment layout (wave-private LDS slice; DS is in-order
    // within a wave so no barrier is needed)
    v16h pf;
    {
      const _Float16* p = &lp[l16 * PSTR + h8];
      v8h lo = *(const v8h*)p;
      v8h hi = *(const v8h*)(p + 16);
#pragma unroll
      for (int i = 0; i < 8; ++i) { pf[i] = lo[i]; pf[8 + i] = hi[i]; }
    }
    // out += P @ V using transposed V (K-contiguous B-fragments)
#pragma unroll
    for (int j = 0; j < 4; ++j) {
      const _Float16* p = vb + (size_t)(j * 16 + l16) * SC_ + kt * 32 + h8;
      v8h lo = *(const v8h*)p;
      v8h hi = *(const v8h*)(p + 16);
      v16h bv;
#pragma unroll
      for (int i = 0; i < 8; ++i) { bv[i] = lo[i]; bv[8 + i] = hi[i]; }
      o[j] = __builtin_amdgcn_wmma_f32_16x16x32_f16(false, pf, false, bv,
                                                    (short)0, o[j], false, false);
    }
  }
  // normalize and store as [B,S,H,Hd] f16 (row-major input for out-proj GEMM)
#pragma unroll
  for (int i = 0; i < 8; ++i) {
    float inv = 1.f / lrow[i];
    int srow = qt * 16 + i + h8;
    _Float16* dst = oh + ((size_t)(b * S_ + srow) * H_ + h) * HD_;
#pragma unroll
    for (int j = 0; j < 4; ++j)
      dst[j * 16 + l16] = (_Float16)(o[j][i] * inv);
  }
}

// ---------------------------------------------------------------------------
extern "C" void kernel_launch(void* const* d_in, const int* in_sizes, int n_in,
                              void* d_out, int out_size, void* d_ws,
                              size_t ws_size, hipStream_t stream) {
  (void)in_sizes; (void)n_in; (void)out_size; (void)ws_size;
  const float* x    = (const float*)d_in[0];
  const float* ctx  = (const float*)d_in[1];
  const float* q_w  = (const float*)d_in[2];
  const float* kv_w = (const float*)d_in[3];
  const float* qn_s = (const float*)d_in[4];
  const float* qn_b = (const float*)d_in[5];
  const float* kn_s = (const float*)d_in[6];
  const float* kn_b = (const float*)d_in[7];
  const float* p_w  = (const float*)d_in[8];
  const float* p_b  = (const float*)d_in[9];
  float* out = (float*)d_out;

  char* ws = (char*)d_ws;
  const size_t MB = 1u << 20;
  _Float16* xh    = (_Float16*)(ws + 0);        // 16 MB
  _Float16* ch    = (_Float16*)(ws + 16 * MB);  // 16 MB
  _Float16* qwh   = (_Float16*)(ws + 32 * MB);  //  2 MB
  _Float16* kvwh  = (_Float16*)(ws + 34 * MB);  //  4 MB
  _Float16* pwh   = (_Float16*)(ws + 38 * MB);  //  2 MB
  float*    qf32  = (float*)   (ws + 40 * MB);  // 32 MB
  float*    kvf32 = (float*)   (ws + 72 * MB);  // 64 MB  (total 136 MB)
  // dead-buffer aliases (stream-serialized, so safe):
  _Float16* qhb = xh;                 // q f16 [B,H,S,Hd]   (xh dead after gemm1)
  _Float16* khb = ch;                 // k f16 [B,H,Sc,Hd]  (ch dead after gemm2)
  _Float16* vtb = (_Float16*)qf32;    // v^T f16 [B,H,Hd,Sc](qf32 dead after ln_q)
  _Float16* ohb = (_Float16*)kvf32;   // attn out f16       (kvf32 dead after ln_kv)

  const int M = B_ * S_;  // 8192

  // 1) fp32 -> fp16 conversions
  ca_f32_to_f16<<<(M * DIM_) / 256, 256, 0, stream>>>(x, xh);
  ca_f32_to_f16<<<(B_ * SC_ * DIM_) / 256, 256, 0, stream>>>(ctx, ch);
  ca_f32_to_f16<<<(DIM_ * DIM_) / 256, 256, 0, stream>>>(q_w, qwh);
  ca_f32_to_f16<<<(DIM_ * 2 * DIM_) / 256, 256, 0, stream>>>(kv_w, kvwh);
  ca_f32_to_f16<<<(DIM_ * DIM_) / 256, 256, 0, stream>>>(p_w, pwh);

  // 2) q / kv projections (WMMA GEMM)
  ca_gemm_f16<<<dim3(M / GBM, DIM_ / GBN), 256, 0, stream>>>(
      xh, qwh, qf32, nullptr, M, DIM_, DIM_);
  ca_gemm_f16<<<dim3(M / GBM, (2 * DIM_) / GBN), 256, 0, stream>>>(
      ch, kvwh, kvf32, nullptr, M, 2 * DIM_, DIM_);

  // 3) per-head LN (+RoPE on q), layout shuffles to WMMA-friendly forms
  ca_ln_rope_q<<<(B_ * S_ * H_) / 8, 256, 0, stream>>>(qf32, qn_s, qn_b, qhb);
  ca_ln_kv<<<(B_ * SC_ * H_) / 8, 256, 0, stream>>>(kvf32, kn_s, kn_b, khb, vtb);

  // 4) flash attention (K/V f16 = 32 MB total -> L2-resident re-streaming)
  ca_flash_attn<<<(B_ * H_ * (S_ / 16)) / 8, 256, 0, stream>>>(qhb, khb, vtb, ohb);

  // 5) output projection + bias (fp32 out)
  ca_gemm_f16<<<dim3(M / GBM, DIM_ / GBN), 256, 0, stream>>>(
      ohb, pwh, out, p_b, M, DIM_, DIM_);
}
